// SGC_33208687133418
// MI455X (gfx1250) — matrix operand order repaired and checked by reading, not compiled, not run on hardware
//
#include <hip/hip_runtime.h>
#include <math.h>

// SGC on MI455X (gfx1250, wave32).
// Phase 1: degree + rsqrt norm (atomics), 2x edge propagation (bandwidth-bound,
//          ~3.5 GB of gather/scatter traffic -> ~150us at 23.3 TB/s).
// Phase 2: WMMA dense transform. Weights pre-packed once into f16 in the exact
//          v_wmma B-operand lane layout (32B contiguous per lane -> pure
//          global_load_b128, no cvt in the hot loop). Hidden layer stored f16.
//          Layer-2 fuses bias + log_softmax in-register via shfl_xor.

typedef __attribute__((ext_vector_type(16))) _Float16 v16h;
typedef __attribute__((ext_vector_type(8)))  float    v8f;

#define IN_F  128
#define OUT_F 64

__global__ void zero_f32(float* __restrict__ p, long n) {
  long i = (long)blockIdx.x * blockDim.x + threadIdx.x;
  if (i < n) p[i] = 0.0f;
}

__global__ void deg_init(float* __restrict__ deg, int n) {
  int i = blockIdx.x * blockDim.x + threadIdx.x;
  if (i < n) deg[i] = 1.0f;  // self-loop contributes 1 to every node's degree
}

__global__ void deg_count(const int* __restrict__ dst, float* __restrict__ deg, int e) {
  int i = blockIdx.x * blockDim.x + threadIdx.x;
  if (i < e) atomicAdd(&deg[dst[i]], 1.0f);
}

__global__ void deg_rsqrt(float* __restrict__ deg, int n) {
  int i = blockIdx.x * blockDim.x + threadIdx.x;
  if (i < n) {
    float d = deg[i];
    deg[i] = (d > 0.0f) ? rsqrtf(d) : 0.0f;
  }
}

// One wave32 per (edge or self-loop); each lane moves 4 contiguous floats.
__global__ void propagate(const float* __restrict__ xin, float* __restrict__ yout,
                          const int* __restrict__ src, const int* __restrict__ dst,
                          const float* __restrict__ dinv, int e, int n) {
  long tid = (long)blockIdx.x * blockDim.x + threadIdx.x;
  long w   = tid >> 5;
  int lane = (int)(tid & 31);
  if (w >= (long)e + n) return;
  int s, d;
  if (w < e) { s = src[w]; d = dst[w]; }
  else       { s = d = (int)(w - e); }
  float nw = dinv[s] * dinv[d];
  const float4 v = *(const float4*)(xin + (size_t)s * IN_F + lane * 4);
  float* o = yout + (size_t)d * IN_F + lane * 4;
  atomicAdd(o + 0, v.x * nw);
  atomicAdd(o + 1, v.y * nw);
  atomicAdd(o + 2, v.z * nw);
  atomicAdd(o + 3, v.w * nw);
}

// Pack row-major f32 W[C][128] into f16 WMMA-B lane layout:
//   flat = ((kb_chunk * ctiles + tile) * 32 + lane) * 16 + i
//   value = W[tile*16 + (lane&15)][kb_chunk*32 + ((lane&16)?8:0) + (i&7) + ((i&8)?16:0)]
// so each lane's 16 halves (32 B) are contiguous -> one v16h load per B tile.
__global__ void pack_w(const float* __restrict__ W, _Float16* __restrict__ P, int ctiles) {
  int idx = blockIdx.x * blockDim.x + threadIdx.x;
  int total = ctiles * 4 * 512;  // 4 K-chunks * 32 lanes * 16 halves per tile
  if (idx >= total) return;
  int i  = idx & 15;
  int l  = (idx >> 4) & 31;
  int t  = (idx >> 9) % ctiles;
  int kb = (idx >> 9) / ctiles;
  int c  = t * 16 + (l & 15);
  int k  = kb * 32 + ((l & 16) ? 8 : 0) + (i & 7) + ((i & 8) ? 16 : 0);
  P[idx] = (_Float16)W[c * IN_F + k];
}

// A-matrix (16x32 f16, MxK) per ISA layout, from f32 source:
//   lane<16:  halves 0..7 = K[kb+0..7],  halves 8..15 = K[kb+16..23]
//   lane>=16: halves 0..7 = K[kb+8..15], halves 8..15 = K[kb+24..31]
__device__ __forceinline__ v16h load_a_f32(const float* __restrict__ X, int rowBase,
                                           int kb, int lane) {
  const int m  = lane & 15;
  const int ko = kb + ((lane & 16) ? 8 : 0);
  const float* p = X + (size_t)(rowBase + m) * IN_F + ko;
  v16h a;
#pragma unroll
  for (int i = 0; i < 8; ++i) {
    a[i]     = (_Float16)p[i];
    a[8 + i] = (_Float16)p[16 + i];
  }
  return a;
}

// Same A layout from an f16 row-major source (two contiguous b128 loads).
__device__ __forceinline__ v16h load_a_f16(const _Float16* __restrict__ X, int rowBase,
                                           int kb, int lane) {
  const int m  = lane & 15;
  const int ko = kb + ((lane & 16) ? 8 : 0);
  const _Float16* p = X + (size_t)(rowBase + m) * IN_F + ko;
  v16h a;
#pragma unroll
  for (int i = 0; i < 8; ++i) {
    a[i]     = p[i];
    a[8 + i] = p[16 + i];
  }
  return a;
}

// Pre-packed B tile: single contiguous 32-byte load per lane.
__device__ __forceinline__ v16h load_bp(const _Float16* __restrict__ P, int ctiles,
                                        int t, int kc, int lane) {
  return *(const v16h*)(P + ((size_t)(kc * ctiles + t) * 512) + lane * 16);
}

// H16 = f16(relu(X @ W1^T + b1)), X:[N,128] f32, W1 packed. Wave per 16-row tile.
__global__ void gemm1_relu(const float* __restrict__ X, const _Float16* __restrict__ W1p,
                           const float* __restrict__ b1, _Float16* __restrict__ H16,
                           int ntiles) {
  const int lane = threadIdx.x & 31;
  const int tile = blockIdx.x * (blockDim.x >> 5) + (threadIdx.x >> 5);
  if (tile >= ntiles) return;  // wave-uniform: EXEC stays all-ones for WMMA
  const int rowBase = tile * 16;

  v8f acc[8] = {};
#pragma unroll
  for (int kc = 0; kc < 4; ++kc) {
    v16h a = load_a_f32(X, rowBase, kc * 32, lane);
#pragma unroll
    for (int t = 0; t < 8; ++t) {
      v16h b = load_bp(W1p, 8, t, kc, lane);
      acc[t] = __builtin_amdgcn_wmma_f32_16x16x32_f16(
          false, a, false, b, (short)0, acc[t], false, false);
    }
  }

  // D layout: VGPR r, lanes 0-15 -> (M=r, N=lane); lanes 16-31 -> (M=8+r, N=lane-16)
  const int n  = lane & 15;
  const int mo = (lane & 16) ? 8 : 0;
#pragma unroll
  for (int t = 0; t < 8; ++t) {
    float bias = b1[t * 16 + n];
#pragma unroll
    for (int r = 0; r < 8; ++r) {
      float v = acc[t][r] + bias;
      H16[(size_t)(rowBase + mo + r) * IN_F + t * 16 + n] =
          (_Float16)((v > 0.0f) ? v : 0.0f);
    }
  }
}

// OUT = log_softmax(H @ W2^T + b2), H:[N,128] f16, W2 packed. Fused epilogue:
// each 16-lane half-group holds complete rows, reduce with shfl_xor {8,4,2,1}.
__global__ void gemm2_lsm(const _Float16* __restrict__ H16, const _Float16* __restrict__ W2p,
                          const float* __restrict__ b2, float* __restrict__ OUT,
                          int ntiles) {
  const int lane = threadIdx.x & 31;
  const int tile = blockIdx.x * (blockDim.x >> 5) + (threadIdx.x >> 5);
  if (tile >= ntiles) return;
  const int rowBase = tile * 16;

  v8f acc[4] = {};
#pragma unroll
  for (int kc = 0; kc < 4; ++kc) {
    v16h a = load_a_f16(H16, rowBase, kc * 32, lane);
#pragma unroll
    for (int t = 0; t < 4; ++t) {
      v16h b = load_bp(W2p, 4, t, kc, lane);
      acc[t] = __builtin_amdgcn_wmma_f32_16x16x32_f16(
          false, a, false, b, (short)0, acc[t], false, false);
    }
  }

  const int n  = lane & 15;
  const int mo = (lane & 16) ? 8 : 0;
  float bias[4];
#pragma unroll
  for (int t = 0; t < 4; ++t) bias[t] = b2[t * 16 + n];

#pragma unroll
  for (int r = 0; r < 8; ++r) {
    float v0 = acc[0][r] + bias[0];
    float v1 = acc[1][r] + bias[1];
    float v2 = acc[2][r] + bias[2];
    float v3 = acc[3][r] + bias[3];
    // row max across 64 logits (4 local + 16-lane group)
    float m = fmaxf(fmaxf(v0, v1), fmaxf(v2, v3));
#pragma unroll
    for (int off = 8; off >= 1; off >>= 1) m = fmaxf(m, __shfl_xor(m, off, 32));
    // row sum of exp
    float s = expf(v0 - m) + expf(v1 - m) + expf(v2 - m) + expf(v3 - m);
#pragma unroll
    for (int off = 8; off >= 1; off >>= 1) s += __shfl_xor(s, off, 32);
    float lg = m + logf(s);
    size_t ob = (size_t)(rowBase + mo + r) * OUT_F + n;
    OUT[ob + 0]  = v0 - lg;
    OUT[ob + 16] = v1 - lg;
    OUT[ob + 32] = v2 - lg;
    OUT[ob + 48] = v3 - lg;
  }
}

extern "C" void kernel_launch(void* const* d_in, const int* in_sizes, int n_in,
                              void* d_out, int out_size, void* d_ws, size_t ws_size,
                              hipStream_t stream) {
  const float* x  = (const float*)d_in[0];
  const int*   ei = (const int*)d_in[1];   // jax default x64-off: int32 on device
  const float* W1 = (const float*)d_in[2];
  const float* b1 = (const float*)d_in[3];
  const float* W2 = (const float*)d_in[4];
  const float* b2 = (const float*)d_in[5];

  const int N = in_sizes[0] / IN_F;   // 100000
  const int E = in_sizes[1] / 2;      // 1600000
  const int* src = ei;
  const int* dst = ei + E;

  // Workspace: dinv[N] | bufA[N*128 f32] | bufB[N*128 f32] | W1p | W2p (~103 MB)
  // H16 (f16) aliases bufA: hop-1 data is dead once hop-2 has been produced.
  char* ws = (char*)d_ws;
  size_t off = 0;
  float* dinv = (float*)(ws + off);
  off += ((size_t)N * sizeof(float) + 255) & ~(size_t)255;
  float* bufA = (float*)(ws + off);
  off += (size_t)N * IN_F * sizeof(float);
  float* bufB = (float*)(ws + off);
  off += (size_t)N * IN_F * sizeof(float);
  _Float16* W1p = (_Float16*)(ws + off);
  off += (size_t)IN_F * IN_F * sizeof(_Float16);
  _Float16* W2p = (_Float16*)(ws + off);
  _Float16* H16 = (_Float16*)bufA;

  const int B = 256;  // 8 waves per block

  // weight pre-pack into WMMA B layout (tiny, L2-resident afterwards)
  pack_w<<<(IN_F * IN_F + B - 1) / B, B, 0, stream>>>(W1, W1p, 8);
  pack_w<<<(OUT_F * IN_F + B - 1) / B, B, 0, stream>>>(W2, W2p, 4);

  // normalization
  deg_init <<<(N + B - 1) / B, B, 0, stream>>>(dinv, N);
  deg_count<<<(E + B - 1) / B, B, 0, stream>>>(dst, dinv, E);
  deg_rsqrt<<<(N + B - 1) / B, B, 0, stream>>>(dinv, N);

  // 2-hop propagation (ping-pong)
  const long nf = (long)N * IN_F;
  const long pthreads = ((long)E + N) * 32;
  zero_f32 <<<(int)((nf + B - 1) / B), B, 0, stream>>>(bufA, nf);
  propagate<<<(int)((pthreads + B - 1) / B), B, 0, stream>>>(x, bufA, src, dst, dinv, E, N);
  zero_f32 <<<(int)((nf + B - 1) / B), B, 0, stream>>>(bufB, nf);
  propagate<<<(int)((pthreads + B - 1) / B), B, 0, stream>>>(bufA, bufB, src, dst, dinv, E, N);

  // dense transform: WMMA gemm1 (+relu, f16 out) then gemm2 (+log_softmax)
  const int ntiles = N / 16;  // 100000 / 16 = 6250 exact
  const int wpb = B / 32;
  const int gblocks = (ntiles + wpb - 1) / wpb;
  gemm1_relu<<<gblocks, B, 0, stream>>>(bufB, W1p, b1, H16, ntiles);
  gemm2_lsm <<<gblocks, B, 0, stream>>>(H16, W2p, b2, (float*)d_out, ntiles);
}
